// Rec_block_59751585022445
// MI455X (gfx1250) — matrix operand order repaired and checked by reading, not compiled, not run on hardware
//
#include <hip/hip_runtime.h>
#include <hip/hip_bf16.h>

// ---------------------------------------------------------------------------
// PatchTST attention block for MI455X (gfx1250, wave32, WMMA 16x16x32 f16)
// B=4, S=2048, D=512, H=16, DK=32
// ---------------------------------------------------------------------------

typedef __attribute__((ext_vector_type(16))) _Float16 v16h;
typedef __attribute__((ext_vector_type(8)))  float    v8f;

#define S_LEN  2048
#define D_MOD  512
#define NH     16
#define DK     32
#define NB     4
#define NROWS  (NB * S_LEN)          // 8192

union AB {
    v16h         v;
    unsigned int u[8];
    _Float16     h[16];
};

union H2 {
    _Float16     h[2];
    unsigned int u;
};

static __device__ __forceinline__ v8f wmma_f16(const v16h& a, const v16h& b, const v8f& c) {
    return __builtin_amdgcn_wmma_f32_16x16x32_f16(false, a, false, b, (short)0, c, false, false);
}

// ------------------------- stage 0: conversions ----------------------------

__global__ void k_cvt_f16(const float* __restrict__ in, _Float16* __restrict__ out, int n) {
    int i = blockIdx.x * 256 + threadIdx.x;
    if (i < n) out[i] = (_Float16)in[i];
}

// Wt[n*512 + k] = (f16) W[k*512 + n]   (512x512)
__global__ void k_wT(const float* __restrict__ W, _Float16* __restrict__ Wt) {
    int i = blockIdx.x * 256 + threadIdx.x;   // 0 .. 262143
    int n = i >> 9;
    int k = i & 511;
    Wt[i] = (_Float16)W[k * D_MOD + n];
}

// ------------------------- stage 1: QKV projection -------------------------
// C[8192,512] = X[8192,512] @ W ; one wave computes a 16x64 C strip:
// per 32-wide k-chunk: 1 A fragment (2xb128) reused across 4 B fragments
// (8xb128) -> 4 WMMAs (2.5 loads/WMMA).
// VMODE 0: store f16 as [b][h][s][dk]   (Q, K)
// VMODE 1: store f16 as [b][h][dk][s]   (V transposed)

template <int VMODE>
__global__ void k_gemm_f16out(const _Float16* __restrict__ X,
                              const _Float16* __restrict__ Wt,
                              const float* __restrict__ bias,
                              _Float16* __restrict__ out) {
    const int lane = threadIdx.x;
    const int qn = lane & 15, hb = lane >> 4;
    const int mt = blockIdx.x;                 // 0..511
    const int nt = blockIdx.y;                 // 0..7  (64-wide N strips)
    const int rowA = mt * 16 + qn;             // A-operand row (M = lane%16)
    const unsigned int* Xu = (const unsigned int*)X;
    const unsigned int* Wu = (const unsigned int*)Wt;

    AB a, b;
    v8f c0 = {}, c1 = {}, c2 = {}, c3 = {};
    v8f* acc[4] = {&c0, &c1, &c2, &c3};

    for (int kc = 0; kc < D_MOD; kc += 32) {
#pragma unroll
        for (int t = 0; t < 8; ++t) {
            int ka = kc + 16 * hb + 2 * t;                           // A layout
            a.u[t] = Xu[(rowA * D_MOD + ka) >> 1];
        }
#pragma unroll
        for (int n = 0; n < 4; ++n) {
            const int col = nt * 64 + n * 16 + qn;                    // B/C column
#pragma unroll
            for (int t = 0; t < 8; ++t) {
                int kb = kc + 16 * hb + 2 * t;                        // B layout
                b.u[t] = Wu[(col * D_MOD + kb) >> 1];
            }
            *acc[n] = wmma_f16(a.v, b.v, *acc[n]);
        }
    }

#pragma unroll
    for (int n = 0; n < 4; ++n) {
        const int col = nt * 64 + n * 16 + qn;
        const float bval = bias[col];
        const int hh = col >> 5, dk = col & 31;
        const v8f& c = *acc[n];
#pragma unroll
        for (int r = 0; r < 8; ++r) {
            int rr = mt * 16 + r + 8 * hb;         // C row (M = r + 8*hi)
            int bb = rr >> 11, ss = rr & (S_LEN - 1);
            float val = c[r] + bval;
            if (VMODE == 0)
                out[(((bb * NH + hh) * S_LEN + ss) * DK) + dk] = (_Float16)val;
            else
                out[(((bb * NH + hh) * DK + dk) * S_LEN) + ss] = (_Float16)val;
        }
    }
}

// ------------------------- stage 2: flash attention ------------------------
// One wave handles one (b,h, 16-query tile). Computes S^T = K_tile x Q^T so
// query index lands in the lane (N) slot: softmax reductions are one
// shfl_xor(16); online-softmax correction is a per-lane scalar.
// Then O^T = V^T x P ; P maps onto the WMMA B layout with one half-exchange.

__global__ void k_attn(const _Float16* __restrict__ Q,
                       const _Float16* __restrict__ K,
                       const _Float16* __restrict__ Vt,
                       _Float16* __restrict__ O,
                       const int* __restrict__ mborder) {
    const int lane = threadIdx.x;
    const int qn = lane & 15, hb = lane >> 4;
    const int qt = blockIdx.x;                 // 0..127 query tile
    const int bh = blockIdx.y;                 // 0..63  (b*16+h)
    const int border = mborder[0];
    const int iRow = qt * 16 + qn;             // global query row (this lane)

    const _Float16* Kbase = K  + (size_t)bh * S_LEN * DK;
    const unsigned int* Qu = (const unsigned int*)(Q  + (size_t)bh * S_LEN * DK);
    const unsigned int* Ku = (const unsigned int*)Kbase;
    const unsigned int* Vu = (const unsigned int*)(Vt + (size_t)bh * DK * S_LEN);

    // Q^T B-operand: lanes 0-15 hold dk 0..7,16..23, lanes 16-31 hold dk 8..15,24..31
    AB bq;
#pragma unroll
    for (int t = 0; t < 8; ++t) {
        int dk = (t < 4 ? 0 : 16) + 8 * hb + 2 * (t & 3);
        bq.u[t] = Qu[(iRow * DK + dk) >> 1];
    }

    const float scale = 0.17677669529663687f;  // 1/sqrt(32)
    float m = -1e30f, l = 0.f;
    v8f accL = {}, accH = {};

    for (int c0 = 0; c0 < S_LEN; c0 += 32) {
        // ---- scores: two 16x16 tiles of S^T -------------------------------
        AB ak;
        v8f s0, s1;
        {
            int key = c0 + qn;
#pragma unroll
            for (int t = 0; t < 8; ++t) {
                int dk = (t < 4 ? 0 : 16) + 8 * hb + 2 * (t & 3);
                ak.u[t] = Ku[(key * DK + dk) >> 1];
            }
            v8f z = {};
            s0 = wmma_f16(ak.v, bq.v, z);
        }
        {
            int key = c0 + 16 + qn;
#pragma unroll
            for (int t = 0; t < 8; ++t) {
                int dk = (t < 4 ? 0 : 16) + 8 * hb + 2 * (t & 3);
                ak.u[t] = Ku[(key * DK + dk) >> 1];
            }
            v8f z = {};
            s1 = wmma_f16(ak.v, bq.v, z);
        }
        if (c0 + 32 < S_LEN)
            __builtin_prefetch((const void*)(Kbase + (c0 + 32 + qn) * DK), 0, 1);

        // ---- scale + additive band bias, row max --------------------------
        float p0[8], p1[8];
        float tmax = -1e30f;
#pragma unroll
        for (int r = 0; r < 8; ++r) {
            int k0 = c0 + r + 8 * hb;          // key of s0[r]
            int k1 = k0 + 16;                   // key of s1[r]
            int d0 = iRow - k0; d0 = d0 < 0 ? -d0 : d0;
            int d1 = iRow - k1; d1 = d1 < 0 ? -d1 : d1;
            float v0 = s0[r] * scale + (d0 <= border ? 1.0f : 0.0f);
            float v1 = s1[r] * scale + (d1 <= border ? 1.0f : 0.0f);
            p0[r] = v0; p1[r] = v1;
            tmax = fmaxf(tmax, fmaxf(v0, v1));
        }
        tmax = fmaxf(tmax, __shfl_xor(tmax, 16, 32));

        // ---- online softmax update ----------------------------------------
        float mNew = fmaxf(m, tmax);
        float corr = __expf(m - mNew);
        float tsum = 0.f;
#pragma unroll
        for (int r = 0; r < 8; ++r) {
            p0[r] = __expf(p0[r] - mNew);
            p1[r] = __expf(p1[r] - mNew);
            tsum += p0[r] + p1[r];
        }
        tsum += __shfl_xor(tsum, 16, 32);
        l = l * corr + tsum;
#pragma unroll
        for (int r = 0; r < 8; ++r) { accL[r] *= corr; accH[r] *= corr; }

        // ---- pack P into B-operand layout (one half-exchange) -------------
        float x0[8], x1[8];
#pragma unroll
        for (int r = 0; r < 8; ++r) {
            x0[r] = __shfl_xor(p0[r], 16, 32);
            x1[r] = __shfl_xor(p1[r], 16, 32);
        }
        AB bp;
#pragma unroll
        for (int j = 0; j < 8; ++j) {
            bp.h[j]     = (_Float16)(hb ? x1[j] : p0[j]);   // keys base+0..7 / 16..23
            bp.h[8 + j] = (_Float16)(hb ? p1[j] : x0[j]);   // keys base+8..15 / 24..31
        }

        // ---- O^T += V^T x P (two 16-dv tiles) -----------------------------
        AB avL, avH;
#pragma unroll
        for (int t = 0; t < 8; ++t) {
            int key = c0 + 16 * hb + 2 * t;
            avL.u[t] = Vu[(qn * S_LEN + key) >> 1];
            avH.u[t] = Vu[((qn + 16) * S_LEN + key) >> 1];
        }
        accL = wmma_f16(avL.v, bp.v, accL);
        accH = wmma_f16(avH.v, bp.v, accH);

        m = mNew;
    }

    // ---- normalize & store O as f16 [row, h*32+dv] ------------------------
    const float rl = 1.0f / l;
    const int bb = bh >> 4, hh = bh & 15;
    const int grow = bb * S_LEN + qt * 16 + qn;
    unsigned int* Ou = (unsigned int*)O;
#pragma unroll
    for (int t = 0; t < 4; ++t) {
        int dv = 8 * hb + 2 * t;
        H2 pl, ph;
        pl.h[0] = (_Float16)(accL[2 * t] * rl);
        pl.h[1] = (_Float16)(accL[2 * t + 1] * rl);
        ph.h[0] = (_Float16)(accH[2 * t] * rl);
        ph.h[1] = (_Float16)(accH[2 * t + 1] * rl);
        Ou[(grow * D_MOD + hh * DK + dv) >> 1]        = pl.u;
        Ou[(grow * D_MOD + hh * DK + dv + 16) >> 1]   = ph.u;
    }
}

// ------------------------- stage 3: output projection ----------------------
// Same 16x64-per-wave tiling; accumulates f32 and writes d_out directly.

__global__ void k_gemm_f32out(const _Float16* __restrict__ X,
                              const _Float16* __restrict__ Wt,
                              const float* __restrict__ bias,
                              float* __restrict__ out) {
    const int lane = threadIdx.x;
    const int qn = lane & 15, hb = lane >> 4;
    const int mt = blockIdx.x, nt = blockIdx.y;
    const int rowA = mt * 16 + qn;
    const unsigned int* Xu = (const unsigned int*)X;
    const unsigned int* Wu = (const unsigned int*)Wt;

    AB a, b;
    v8f c0 = {}, c1 = {}, c2 = {}, c3 = {};
    v8f* acc[4] = {&c0, &c1, &c2, &c3};

    for (int kc = 0; kc < D_MOD; kc += 32) {
#pragma unroll
        for (int t = 0; t < 8; ++t) {
            int ka = kc + 16 * hb + 2 * t;
            a.u[t] = Xu[(rowA * D_MOD + ka) >> 1];
        }
#pragma unroll
        for (int n = 0; n < 4; ++n) {
            const int col = nt * 64 + n * 16 + qn;
#pragma unroll
            for (int t = 0; t < 8; ++t) {
                int kb = kc + 16 * hb + 2 * t;
                b.u[t] = Wu[(col * D_MOD + kb) >> 1];
            }
            *acc[n] = wmma_f16(a.v, b.v, *acc[n]);
        }
    }

#pragma unroll
    for (int n = 0; n < 4; ++n) {
        const int col = nt * 64 + n * 16 + qn;
        const float bval = bias[col];
        const v8f& c = *acc[n];
#pragma unroll
        for (int r = 0; r < 8; ++r) {
            int rr = mt * 16 + r + 8 * hb;
            out[rr * D_MOD + col] = c[r] + bval;
        }
    }
}

// ---------------------------------------------------------------------------

extern "C" void kernel_launch(void* const* d_in, const int* in_sizes, int n_in,
                              void* d_out, int out_size, void* d_ws, size_t ws_size,
                              hipStream_t stream) {
    const float* x  = (const float*)d_in[0];
    const float* Wq = (const float*)d_in[1];
    const float* bq = (const float*)d_in[2];
    const float* Wk = (const float*)d_in[3];
    const float* bk = (const float*)d_in[4];
    const float* Wv = (const float*)d_in[5];
    const float* bv = (const float*)d_in[6];
    const float* Wo = (const float*)d_in[7];
    const float* bo = (const float*)d_in[8];
    const int*   mb = (const int*)d_in[9];

    char* ws = (char*)d_ws;
    size_t off = 0;
    const size_t SZ_X = (size_t)NROWS * D_MOD * sizeof(_Float16);   // 8 MB
    const size_t SZ_W = (size_t)D_MOD * D_MOD * sizeof(_Float16);   // 512 KB

    _Float16* Xh  = (_Float16*)(ws + off); off += SZ_X;
    _Float16* WqT = (_Float16*)(ws + off); off += SZ_W;
    _Float16* WkT = (_Float16*)(ws + off); off += SZ_W;
    _Float16* WvT = (_Float16*)(ws + off); off += SZ_W;
    _Float16* WoT = (_Float16*)(ws + off); off += SZ_W;
    _Float16* Qh  = (_Float16*)(ws + off); off += SZ_X;
    _Float16* Kh  = (_Float16*)(ws + off); off += SZ_X;
    _Float16* Vth = (_Float16*)(ws + off); off += SZ_X;
    _Float16* Oh  = (_Float16*)(ws + off); off += SZ_X;

    const int nX = NROWS * D_MOD;                       // 4,194,304
    k_cvt_f16<<<nX / 256, 256, 0, stream>>>(x, Xh, nX);
    k_wT<<<(D_MOD * D_MOD) / 256, 256, 0, stream>>>(Wq, WqT);
    k_wT<<<(D_MOD * D_MOD) / 256, 256, 0, stream>>>(Wk, WkT);
    k_wT<<<(D_MOD * D_MOD) / 256, 256, 0, stream>>>(Wv, WvT);
    k_wT<<<(D_MOD * D_MOD) / 256, 256, 0, stream>>>(Wo, WoT);

    dim3 gg(NROWS / 16, D_MOD / 64), bbk(32);
    k_gemm_f16out<0><<<gg, bbk, 0, stream>>>(Xh, WqT, bq, Qh);
    k_gemm_f16out<0><<<gg, bbk, 0, stream>>>(Xh, WkT, bk, Kh);
    k_gemm_f16out<1><<<gg, bbk, 0, stream>>>(Xh, WvT, bv, Vth);

    k_attn<<<dim3(S_LEN / 16, NB * NH), bbk, 0, stream>>>(Qh, Kh, Vth, Oh, mb);

    k_gemm_f32out<<<gg, bbk, 0, stream>>>(Oh, WoT, bo, (float*)d_out);
}
